// EdgeUpdate_20959440404595
// MI455X (gfx1250) — matrix operand order repaired and checked by reading, not compiled, not run on hardware
//
#include <hip/hip_runtime.h>

typedef __attribute__((ext_vector_type(16))) __bf16 v16bf;
typedef __attribute__((ext_vector_type(8)))  __bf16 v8bf;
typedef __attribute__((ext_vector_type(4)))  __bf16 v4bf;
typedef __attribute__((ext_vector_type(8)))  float  v8f;

#define NEDGES   640000
#define NDIM     128      // node feature dim
#define EIN      128      // edge feature dim
#define KDIM     256      // NDIM + EIN
#define ODIM     128      // output dim
#define WS2      264      // bf16 row stride: 256 + 8 pad (16B align, conflict-free)
#define WAVES    4
#define TILE_M   32       // edges per wave-iteration (two 16-row WMMA M-tiles)

// LDS (bf16): [W_hi 128xWS2][W_lo 128xWS2][per-wave: A_hi 32xWS2, A_lo 32xWS2]
#define OFF_WLO   (ODIM * WS2)               // 33792
#define OFF_A     (2 * ODIM * WS2)           // 67584
#define A_SLAB    (2 * TILE_M * WS2)         // 16896 per wave
#define LDS_BF16  (OFF_A + WAVES * A_SLAB)   // 135168 bf16 = 264 KB

union Frag { v16bf v; v8bf h[2]; };

__device__ __forceinline__ void split4(const float4 w, v4bf& hi, v4bf& lo) {
    hi[0] = (__bf16)w.x; lo[0] = (__bf16)(w.x - (float)hi[0]);
    hi[1] = (__bf16)w.y; lo[1] = (__bf16)(w.y - (float)hi[1]);
    hi[2] = (__bf16)w.z; lo[2] = (__bf16)(w.z - (float)hi[2]);
    hi[3] = (__bf16)w.w; lo[3] = (__bf16)(w.w - (float)hi[3]);
}

__global__ __launch_bounds__(128, 1)
void edge_update_wmma_bf16x3(const float* __restrict__ nf,      // [N, 128]
                             const long long* __restrict__ ei,  // [2, E]
                             const float* __restrict__ ef,      // [E, 128]
                             const float* __restrict__ W,       // [128, 256]
                             const float* __restrict__ bias,    // [128]
                             float* __restrict__ out)           // [E, 128]
{
    __shared__ __bf16 smem[LDS_BF16];

    const int tid   = threadIdx.x;
    const int wave  = tid >> 5;
    const int lane  = tid & 31;
    const int lrow  = lane & 15;   // M row (A) or N col (B/C)
    const int khalf = lane >> 4;

    // ---- Stage W into LDS as hi/lo bf16 split ----
    for (int i = tid; i < ODIM * (KDIM / 4); i += blockDim.x) {
        const int row = i >> 6;
        const int c4  = (i & 63) << 2;
        const float4 w = *(const float4*)(W + (size_t)row * KDIM + c4);
        v4bf hi, lo;
        split4(w, hi, lo);
        *(v4bf*)(smem + row * WS2 + c4)           = hi;
        *(v4bf*)(smem + OFF_WLO + row * WS2 + c4) = lo;
    }
    __syncthreads();

    __bf16* Ahi = smem + OFF_A + wave * A_SLAB;
    __bf16* Alo = Ahi + TILE_M * WS2;

    float bval[8];
#pragma unroll
    for (int n = 0; n < 8; ++n) bval[n] = bias[n * 16 + lrow];

    const int nTiles  = NEDGES / TILE_M;   // 20000
    const int tstride = gridDim.x * WAVES;
    for (int tile = blockIdx.x * WAVES + wave; tile < nTiles; tile += tstride) {
        const int e0 = tile * TILE_M;

        // ---- Prefetch next tile's gather targets + streaming inputs ----
        const int ne0 = e0 + tstride * TILE_M;
        if (ne0 < NEDGES) {
            // next edge indices: lane -> edge ne0+lane (src and dst)
            const long long ps = ei[ne0 + lane];
            const long long pd = ei[NEDGES + ne0 + lane];
            const float* prs = nf + (size_t)ps * NDIM;
            const float* prd = nf + (size_t)pd * NDIM;
#pragma unroll
            for (int c = 0; c < 4; ++c) {            // 512B row = 4 cachelines
                __builtin_prefetch(prs + c * 32, 0, 3);
                __builtin_prefetch(prd + c * 32, 0, 3);
            }
            // next edge features: 32 rows x 512B, one row per lane
            const float* pre = ef + (size_t)(ne0 + lane) * EIN;
#pragma unroll
            for (int c = 0; c < 4; ++c) __builtin_prefetch(pre + c * 32, 0, 3);
        }

        // ---- Build A tile: pair = nf[src]+nf[dst] | ef, split into hi/lo bf16 ----
        for (int e = 0; e < TILE_M; ++e) {
            const long long s = ei[e0 + e];
            const long long d = ei[NEDGES + e0 + e];
            const float4 a = *(const float4*)(nf + (size_t)s * NDIM + lane * 4);
            const float4 b = *(const float4*)(nf + (size_t)d * NDIM + lane * 4);
            float4 p;
            p.x = a.x + b.x; p.y = a.y + b.y; p.z = a.z + b.z; p.w = a.w + b.w;
            v4bf hi, lo;
            split4(p, hi, lo);
            *(v4bf*)(Ahi + e * WS2 + lane * 4) = hi;
            *(v4bf*)(Alo + e * WS2 + lane * 4) = lo;
            const float4 ev = *(const float4*)(ef + (size_t)(e0 + e) * EIN + lane * 4);
            split4(ev, hi, lo);
            *(v4bf*)(Ahi + e * WS2 + NDIM + lane * 4) = hi;
            *(v4bf*)(Alo + e * WS2 + NDIM + lane * 4) = lo;
        }

        // ---- GEMM: C[32x128] = A[32x256] * W^T, bf16 16x16x32 WMMA, BF16x3 ----
        // Each B fragment pair feeds 6 WMMAs (2 M-tiles x 3 split terms).
        v8f acc[2][8] = {};
        const __bf16* ahR0 = Ahi + lrow * WS2 + khalf * 8;          // M-tile 0
        const __bf16* alR0 = Alo + lrow * WS2 + khalf * 8;
        const __bf16* ahR1 = ahR0 + 16 * WS2;                       // M-tile 1
        const __bf16* alR1 = alR0 + 16 * WS2;
        const __bf16* bhRow = smem + lrow * WS2 + khalf * 8;
        const __bf16* blRow = smem + OFF_WLO + lrow * WS2 + khalf * 8;

        for (int kc = 0; kc < KDIM; kc += 32) {
            Frag ah0, al0, ah1, al1;
            ah0.h[0] = *(const v8bf*)(ahR0 + kc);
            ah0.h[1] = *(const v8bf*)(ahR0 + kc + 16);
            al0.h[0] = *(const v8bf*)(alR0 + kc);
            al0.h[1] = *(const v8bf*)(alR0 + kc + 16);
            ah1.h[0] = *(const v8bf*)(ahR1 + kc);
            ah1.h[1] = *(const v8bf*)(ahR1 + kc + 16);
            al1.h[0] = *(const v8bf*)(alR1 + kc);
            al1.h[1] = *(const v8bf*)(alR1 + kc + 16);
#pragma unroll
            for (int n = 0; n < 8; ++n) {
                Frag bh, bl;
                bh.h[0] = *(const v8bf*)(bhRow + n * 16 * WS2 + kc);
                bh.h[1] = *(const v8bf*)(bhRow + n * 16 * WS2 + kc + 16);
                bl.h[0] = *(const v8bf*)(blRow + n * 16 * WS2 + kc);
                bl.h[1] = *(const v8bf*)(blRow + n * 16 * WS2 + kc + 16);
                acc[0][n] = __builtin_amdgcn_wmma_f32_16x16x32_bf16(
                    false, al0.v, false, bh.v, (short)0, acc[0][n], false, false);
                acc[1][n] = __builtin_amdgcn_wmma_f32_16x16x32_bf16(
                    false, al1.v, false, bh.v, (short)0, acc[1][n], false, false);
                acc[0][n] = __builtin_amdgcn_wmma_f32_16x16x32_bf16(
                    false, ah0.v, false, bl.v, (short)0, acc[0][n], false, false);
                acc[1][n] = __builtin_amdgcn_wmma_f32_16x16x32_bf16(
                    false, ah1.v, false, bl.v, (short)0, acc[1][n], false, false);
                acc[0][n] = __builtin_amdgcn_wmma_f32_16x16x32_bf16(
                    false, ah0.v, false, bh.v, (short)0, acc[0][n], false, false);
                acc[1][n] = __builtin_amdgcn_wmma_f32_16x16x32_bf16(
                    false, ah1.v, false, bh.v, (short)0, acc[1][n], false, false);
            }
        }

        // ---- Epilogue: bias + ReLU ----
        // C layout: VGPR i -> M = mt*16 + i + khalf*8, N = lrow
#pragma unroll
        for (int mt = 0; mt < 2; ++mt) {
#pragma unroll
            for (int n = 0; n < 8; ++n) {
                const float bb = bval[n];
#pragma unroll
                for (int i = 0; i < 8; ++i) {
                    const int m = mt * 16 + i + khalf * 8;
                    float v = acc[mt][n][i] + bb;
                    v = v > 0.0f ? v : 0.0f;
                    out[(size_t)(e0 + m) * ODIM + n * 16 + lrow] = v;
                }
            }
        }
    }
}

extern "C" void kernel_launch(void* const* d_in, const int* in_sizes, int n_in,
                              void* d_out, int out_size, void* d_ws, size_t ws_size,
                              hipStream_t stream) {
    (void)in_sizes; (void)n_in; (void)d_ws; (void)ws_size; (void)out_size;
    const float*     nf = (const float*)d_in[0];
    const long long* ei = (const long long*)d_in[1];
    const float*     ef = (const float*)d_in[2];
    const float*     W  = (const float*)d_in[3];
    const float*     b  = (const float*)d_in[4];
    float*           o  = (float*)d_out;

    const int blocks = 512;  // 4 waves each; ~10 tiles/wave amortizes W staging
    edge_update_wmma_bf16x3<<<blocks, 128, 0, stream>>>(nf, ei, ef, W, b, o);
}
